// CausalSelfAttention_32040456028776
// MI455X (gfx1250) — matrix operand order repaired and checked by reading, not compiled, not run on hardware
//
#include <hip/hip_runtime.h>
#include <cstdint>
#include <cstddef>

// Problem constants (from reference)
#define B_  2
#define S_  2048
#define D_  2048
#define H_  16
#define HD_ 128
#define M_  (B_*S_)   // 4096 rows of x / attn
#define K_  D_
#define N_  D_

typedef __attribute__((ext_vector_type(16))) __bf16 bf16x16;
typedef __attribute__((ext_vector_type(8)))  float  f32x8;

#define WMMA_BF16(a,b,c) \
  __builtin_amdgcn_wmma_f32_16x16x32_bf16(false,(a),false,(b),(short)0,(c),false,false)

__device__ __forceinline__ unsigned short f2bf(float x) {
  unsigned u = __float_as_uint(x);
  unsigned r = (u + 0x7FFFu + ((u >> 16) & 1u)) >> 16;   // round-to-nearest-even
  return (unsigned short)r;
}

__device__ __forceinline__ f32x8 f8zero() {
  f32x8 z = {0.f,0.f,0.f,0.f,0.f,0.f,0.f,0.f};
  return z;
}

// Load a 32-byte bf16 fragment from two 16-byte chunks.
__device__ __forceinline__ bf16x16 frag2(const unsigned short* p0, const unsigned short* p1) {
  union { uint4 q[2]; bf16x16 v; } u;
  u.q[0] = *(const uint4*)p0;
  u.q[1] = *(const uint4*)p1;
  return u.v;
}

// ---------------------------------------------------------------- converts
__global__ __launch_bounds__(256)
void k_cvt(const float* __restrict__ in, unsigned short* __restrict__ out, int n) {
  int i = blockIdx.x * 256 + threadIdx.x;
  if (i < n) out[i] = f2bf(in[i]);
}

// wT[n][k] = w[k][n] (bf16); output-linear so writes are coalesced.
__global__ __launch_bounds__(256)
void k_cvt_T(const float* __restrict__ w, unsigned short* __restrict__ wT) {
  int i = blockIdx.x * 256 + threadIdx.x;   // i = n*K_ + k
  int k = i & (K_ - 1);
  int n = i >> 11;
  wT[i] = f2bf(w[(size_t)k * N_ + n]);
}

// V [B,H,S,HD] -> Vt [B,H,HD,S]
__global__ __launch_bounds__(256)
void k_transpose(const unsigned short* __restrict__ V, unsigned short* __restrict__ Vt) {
  int i  = blockIdx.x * 256 + threadIdx.x;  // over B*H*S*HD
  int hd = i & 127;
  int s  = (i >> 7) & 2047;
  int bh = i >> 18;
  Vt[((size_t)bh * HD_ + hd) * S_ + s] = V[i];
}

// ---------------------------------------------------------------- GEMM
// C[M_,N_] = A[M_,K_] * B[K_,N_], A bf16 row-major, Bt = B^T bf16 [N_,K_].
// MODE 0: write bf16 into [B,H,S,HD] head-major layout (for Q/K/V)
// MODE 1: write fp32 row-major [M_,N_] (final output)
// Block tile 256(M) x 128(N), 8 waves in 4x2 grid, wave tile 64x64
// (16 WMMA per 8 fragment loads per K-step), K-step 32, double-buffered LDS
// with register-staged global loads overlapping the WMMAs of the prior tile.
template<int MODE>
__global__ __launch_bounds__(256)
void k_gemm(const unsigned short* __restrict__ A,
            const unsigned short* __restrict__ Bt,
            void* __restrict__ Cout)
{
  __shared__ unsigned short lA[2][256 * 32];   // [buf][row][k]   32KB
  __shared__ unsigned short lB[2][128 * 32];   // [buf][n][k]     16KB

  const int mBlock = blockIdx.y * 256;
  const int nBlock = blockIdx.x * 128;
  const int t    = threadIdx.x;
  const int wave = t >> 5;                  // 8 waves
  const int lane = t & 31;
  const int half = lane >> 4;
  const int lr   = lane & 15;
  const int wm   = (wave & 3) * 64;         // 4 waves over M
  const int wn   = (wave >> 2) * 64;        // 2 waves over N

  // cooperative loaders: A -> one 32-elem row per thread; B -> half row
  const unsigned short* gA = A  + (size_t)(mBlock + t) * K_;
  const unsigned short* gB = Bt + (size_t)(nBlock + (t >> 1)) * K_ + (t & 1) * 16;
  const int aOff = t * 32;
  const int bOff = (t >> 1) * 32 + (t & 1) * 16;

  f32x8 acc[4][4];
  #pragma unroll
  for (int i = 0; i < 4; ++i)
    #pragma unroll
    for (int j = 0; j < 4; ++j) acc[i][j] = f8zero();

  // prologue: stage tile 0 and commit to buffer 0
  uint4 ra0 = *(const uint4*)(gA);
  uint4 ra1 = *(const uint4*)(gA + 8);
  uint4 ra2 = *(const uint4*)(gA + 16);
  uint4 ra3 = *(const uint4*)(gA + 24);
  uint4 rb0 = *(const uint4*)(gB);
  uint4 rb1 = *(const uint4*)(gB + 8);
  int buf = 0;
  *(uint4*)(&lA[0][aOff])      = ra0;
  *(uint4*)(&lA[0][aOff + 8])  = ra1;
  *(uint4*)(&lA[0][aOff + 16]) = ra2;
  *(uint4*)(&lA[0][aOff + 24]) = ra3;
  *(uint4*)(&lB[0][bOff])      = rb0;
  *(uint4*)(&lB[0][bOff + 8])  = rb1;
  __syncthreads();

  const int NK = K_ / 32;
  for (int kt = 0; kt < NK; ++kt) {
    // stage next tile while computing this one
    if (kt + 1 < NK) {
      const int k0 = (kt + 1) * 32;
      ra0 = *(const uint4*)(gA + k0);
      ra1 = *(const uint4*)(gA + k0 + 8);
      ra2 = *(const uint4*)(gA + k0 + 16);
      ra3 = *(const uint4*)(gA + k0 + 24);
      rb0 = *(const uint4*)(gB + k0);
      rb1 = *(const uint4*)(gB + k0 + 8);
    }

    bf16x16 af[4], bfr[4];
    // A fragment: lane lr holds row, K split {0..7,16..23} / {8..15,24..31}
    #pragma unroll
    for (int mt = 0; mt < 4; ++mt) {
      const unsigned short* p = &lA[buf][(wm + mt * 16 + lr) * 32 + half * 8];
      af[mt] = frag2(p, p + 16);
    }
    // B fragment: lane lr holds column, K split {0..7,16..23} / {8..15,24..31}
    #pragma unroll
    for (int nt = 0; nt < 4; ++nt) {
      const unsigned short* p = &lB[buf][(wn + nt * 16 + lr) * 32 + half * 8];
      bfr[nt] = frag2(p, p + 16);
    }
    #pragma unroll
    for (int mt = 0; mt < 4; ++mt)
      #pragma unroll
      for (int nt = 0; nt < 4; ++nt)
        acc[mt][nt] = WMMA_BF16(af[mt], bfr[nt], acc[mt][nt]);

    if (kt + 1 < NK) {
      __syncthreads();              // everyone done reading buf^1 (iter kt-1)
      buf ^= 1;
      *(uint4*)(&lA[buf][aOff])      = ra0;
      *(uint4*)(&lA[buf][aOff + 8])  = ra1;
      *(uint4*)(&lA[buf][aOff + 16]) = ra2;
      *(uint4*)(&lA[buf][aOff + 24]) = ra3;
      *(uint4*)(&lB[buf][bOff])      = rb0;
      *(uint4*)(&lB[buf][bOff + 8])  = rb1;
      __syncthreads();              // stores visible before next reads
    }
  }

  // Epilogue. C layout: VGPR j, lanes 0-15 -> (M=j, N=lr); lanes 16-31 -> (M=j+8, N=lr)
  #pragma unroll
  for (int mt = 0; mt < 4; ++mt) {
    const int m0 = mBlock + wm + mt * 16 + half * 8;
    #pragma unroll
    for (int nt = 0; nt < 4; ++nt) {
      f32x8 c = acc[mt][nt];
      const int n = nBlock + wn + nt * 16 + lr;
      if (MODE == 0) {
        const int b = m0 >> 11, s0 = m0 & 2047;
        const int h = n >> 7,  hd = n & 127;
        unsigned short* p =
            (unsigned short*)Cout + ((size_t)(b * H_ + h) * S_ + s0) * HD_ + hd;
        #pragma unroll
        for (int j = 0; j < 8; ++j) p[(size_t)j * HD_] = f2bf(c[j]);
      } else {
        float* p = (float*)Cout + (size_t)m0 * N_ + n;
        #pragma unroll
        for (int j = 0; j < 8; ++j) p[(size_t)j * N_] = c[j];
      }
    }
  }
}

// ---------------------------------------------------------------- flash attention
// Q,K: [B,H,S,HD] bf16; Vt: [B,H,HD,S] bf16; O: [B,S,D] bf16 (head-concat)
__global__ __launch_bounds__(256)
void k_attn(const unsigned short* __restrict__ Q,
            const unsigned short* __restrict__ Kmat,
            const unsigned short* __restrict__ Vt,
            unsigned short* __restrict__ O)
{
  __shared__ unsigned short lp[8][16 * 32];   // per-wave P tile (16q x 32k)

  const int bh   = blockIdx.x >> 4;           // S_/128 = 16 query blocks
  const int qblk = blockIdx.x & 15;
  const int wave = threadIdx.x >> 5;
  const int lane = threadIdx.x & 31;
  const int half = lane >> 4;
  const int lr   = lane & 15;
  const int q0   = qblk * 128 + wave * 16;    // 16 queries per wave

  const unsigned short* Qb = Q    + (size_t)bh * S_ * HD_;
  const unsigned short* Kb = Kmat + (size_t)bh * S_ * HD_;
  const unsigned short* Vb = Vt   + (size_t)bh * HD_ * S_;
  unsigned short* myp = lp[wave];

  // Q fragments (A layout), HD=128 -> 4 chunks of K=32
  bf16x16 qf[4];
  #pragma unroll
  for (int c = 0; c < 4; ++c) {
    const unsigned short* p = Qb + (size_t)(q0 + lr) * HD_ + c * 32 + half * 8;
    qf[c] = frag2(p, p + 16);
  }

  f32x8 o[8];
  float m_i[8], l_i[8];
  #pragma unroll
  for (int t8 = 0; t8 < 8; ++t8) o[t8] = f8zero();
  #pragma unroll
  for (int j = 0; j < 8; ++j) { m_i[j] = -__builtin_inff(); l_i[j] = 0.f; }

  const float scale = 0.08838834764831845f;   // 1/sqrt(HD)
  const int nkc = (q0 + 15) / 32 + 1;         // causal: key chunks of 32

  for (int kc = 0; kc < nkc; ++kc) {
    const int key0 = kc * 32;

    // S = Q K^T  (16 x 32 scores = two 16x16 tiles)
    f32x8 sc[2];
    sc[0] = f8zero(); sc[1] = f8zero();
    #pragma unroll
    for (int nt = 0; nt < 2; ++nt) {
      #pragma unroll
      for (int c = 0; c < 4; ++c) {
        const unsigned short* p =
            Kb + (size_t)(key0 + nt * 16 + lr) * HD_ + c * 32 + half * 8;
        bf16x16 kf = frag2(p, p + 16);           // B frag: column = key row
        sc[nt] = WMMA_BF16(qf[c], kf, sc[nt]);
      }
    }

    // scale + causal mask (only chunks touching the diagonal)
    const bool edge = (key0 + 31 > q0);
    #pragma unroll
    for (int nt = 0; nt < 2; ++nt) {
      int n_g = key0 + nt * 16 + lr;
      #pragma unroll
      for (int j = 0; j < 8; ++j) {
        float v = sc[nt][j] * scale;
        if (edge && (n_g > q0 + j + half * 8)) v = -__builtin_inff();
        sc[nt][j] = v;
      }
    }

    // online softmax: per-row max / sum via intra-half-wave xor reductions
    float alpha[8];
    #pragma unroll
    for (int j = 0; j < 8; ++j) {
      float v = fmaxf(sc[0][j], sc[1][j]);
      v = fmaxf(v, __shfl_xor(v, 1, 32));
      v = fmaxf(v, __shfl_xor(v, 2, 32));
      v = fmaxf(v, __shfl_xor(v, 4, 32));
      v = fmaxf(v, __shfl_xor(v, 8, 32));
      float mnew = fmaxf(m_i[j], v);
      alpha[j] = __expf(m_i[j] - mnew);
      m_i[j] = mnew;
    }
    #pragma unroll
    for (int nt = 0; nt < 2; ++nt)
      #pragma unroll
      for (int j = 0; j < 8; ++j)
        sc[nt][j] = __expf(sc[nt][j] - m_i[j]);
    #pragma unroll
    for (int j = 0; j < 8; ++j) {
      float v = sc[0][j] + sc[1][j];
      v += __shfl_xor(v, 1, 32);
      v += __shfl_xor(v, 2, 32);
      v += __shfl_xor(v, 4, 32);
      v += __shfl_xor(v, 8, 32);
      l_i[j] = l_i[j] * alpha[j] + v;
    }
    #pragma unroll
    for (int t8 = 0; t8 < 8; ++t8)
      #pragma unroll
      for (int j = 0; j < 8; ++j)
        o[t8][j] *= alpha[j];

    // re-layout P: C layout -> LDS -> A layout (same-wave DS ops are ordered)
    #pragma unroll
    for (int nt = 0; nt < 2; ++nt)
      #pragma unroll
      for (int j = 0; j < 8; ++j)
        myp[(j + half * 8) * 32 + nt * 16 + lr] = f2bf(sc[nt][j]);
    asm volatile("s_wait_dscnt 0" ::: "memory");
    const unsigned short* pp = myp + lr * 32 + half * 8;
    bf16x16 pf = frag2(pp, pp + 16);

    // O += P @ V  (Vt rows are hd, contiguous over keys)
    #pragma unroll
    for (int t8 = 0; t8 < 8; ++t8) {
      const unsigned short* pv = Vb + (size_t)(t8 * 16 + lr) * S_ + key0 + half * 8;
      bf16x16 vf = frag2(pv, pv + 16);
      o[t8] = WMMA_BF16(pf, vf, o[t8]);
    }
  }

  // normalize + store bf16 into [B,S,D] with head concat
  const int b = bh >> 4, h = bh & 15;
  #pragma unroll
  for (int j = 0; j < 8; ++j) {
    float inv = 1.0f / l_i[j];
    int srow = q0 + j + half * 8;
    size_t base = ((size_t)b * S_ + srow) * (size_t)D_ + h * HD_;
    #pragma unroll
    for (int t8 = 0; t8 < 8; ++t8)
      O[base + t8 * 16 + lr] = f2bf(o[t8][j] * inv);
  }
}

// ---------------------------------------------------------------- launch
extern "C" void kernel_launch(void* const* d_in, const int* in_sizes, int n_in,
                              void* d_out, int out_size, void* d_ws, size_t ws_size,
                              hipStream_t stream)
{
  const float* x  = (const float*)d_in[0];
  const float* wq = (const float*)d_in[1];
  const float* wk = (const float*)d_in[2];
  const float* wv = (const float*)d_in[3];
  const float* wo = (const float*)d_in[4];

  char* ws = (char*)d_ws;
  const size_t MB = 1024 * 1024;
  unsigned short* xb  = (unsigned short*)(ws + 0 * MB);    // x bf16        16MB
  unsigned short* wqT = (unsigned short*)(ws + 16 * MB);   // wq^T bf16      8MB
  unsigned short* wkT = (unsigned short*)(ws + 24 * MB);   // wk^T bf16      8MB
  unsigned short* wvT = (unsigned short*)(ws + 32 * MB);   // wv^T bf16      8MB
  unsigned short* woT = (unsigned short*)(ws + 40 * MB);   // wo^T bf16      8MB
  unsigned short* Qb  = (unsigned short*)(ws + 48 * MB);   // Q [B,H,S,HD]  16MB
  unsigned short* Kb  = (unsigned short*)(ws + 64 * MB);   // K [B,H,S,HD]  16MB
  unsigned short* Vb  = (unsigned short*)(ws + 80 * MB);   // V [B,H,S,HD]  16MB
  unsigned short* VtB = (unsigned short*)(ws + 96 * MB);   // V^T [B,H,HD,S]16MB
  unsigned short* Ab  = (unsigned short*)(ws + 112 * MB);  // attn [B,S,D]  16MB

  const int nX = M_ * K_;        // 8388608
  const int nW = D_ * D_;        // 4194304

  k_cvt<<<nX / 256, 256, 0, stream>>>(x, xb, nX);
  k_cvt_T<<<nW / 256, 256, 0, stream>>>(wq, wqT);
  k_cvt_T<<<nW / 256, 256, 0, stream>>>(wk, wkT);
  k_cvt_T<<<nW / 256, 256, 0, stream>>>(wv, wvT);
  k_cvt_T<<<nW / 256, 256, 0, stream>>>(wo, woT);

  dim3 g(N_ / 128, M_ / 256);   // 16 x 16
  k_gemm<0><<<g, 256, 0, stream>>>(xb, wqT, Qb);
  k_gemm<0><<<g, 256, 0, stream>>>(xb, wkT, Kb);
  k_gemm<0><<<g, 256, 0, stream>>>(xb, wvT, Vb);

  k_transpose<<<nX / 256, 256, 0, stream>>>(Vb, VtB);

  k_attn<<<B_ * H_ * (S_ / 128), 256, 0, stream>>>(Qb, Kb, VtB, Ab);

  k_gemm<1><<<g, 256, 0, stream>>>(Ab, woT, d_out);
}
